// GCNClassifier_56727928045765
// MI455X (gfx1250) — matrix operand order repaired and checked by reading, not compiled, not run on hardware
//
#include <hip/hip_runtime.h>
#include <hip/hip_bf16.h>

// GCN classifier for MI455X (gfx1250, wave32, WMMA).
//
// Pipeline per call:
//   for layer in {1,2,3}:
//     agg = 0
//     agg[dst] += h[src]            (L2-resident scatter-add, f32 atomics)
//     h = elu(agg @ W + b)          (WMMA f16 x f16 -> f32 accumulate)
//   sums[g] += h[node], counts[g]++  -> hg = sums/max(counts,1)
//   out = hg @ Wc + bc
//
// Workspace layout (bytes):
//   [0,            32MB)  agg   (65536 x 128 f32)
//   [32MB,         64MB)  h     (65536 x 128 f32)
//   [64MB, +3*32KB)       WT1..WT3 (128x128 f16, transposed: WT[n*128+k])
//   then sums (64x128 f32), counts (64 f32)

#define HID 128
#define NNODES 65536
#define NEDGES 524288
#define NGRAPHS 64
#define OUTD 10

typedef __attribute__((ext_vector_type(16))) _Float16 v16h;
typedef __attribute__((ext_vector_type(8)))  float    v8f;

// ---------------------------------------------------------------------------
// Convert W (128x128 f32, row-major [k][n]) -> WT (f16, [n][k]) so that a B
// fragment (32x16 f16, lane n = L%16, K = kbase + (L/16)*16 + h) is one
// contiguous 32-byte per-lane load.
// ---------------------------------------------------------------------------
__global__ void convert_wt_kernel(const float* __restrict__ W,
                                  _Float16* __restrict__ WT) {
  int k = blockIdx.x;    // 0..127
  int n = threadIdx.x;   // 0..127
  WT[n * HID + k] = (_Float16)W[k * HID + n];
}

// ---------------------------------------------------------------------------
// Edge scatter-add: one wave per edge; lane covers 4 contiguous dims.
// H and agg are both 32MB -> L2-resident on MI455X (192MB L2).
// ---------------------------------------------------------------------------
__global__ void scatter_add_kernel(const float* __restrict__ H,
                                   const int* __restrict__ src,
                                   const int* __restrict__ dst,
                                   float* __restrict__ agg) {
  int wavesPerBlock = blockDim.x >> 5;
  int e = blockIdx.x * wavesPerBlock + (threadIdx.x >> 5);
  if (e >= NEDGES) return;
  int lane = threadIdx.x & 31;
  int s = src[e];
  int d = dst[e];
  const float4* hs = (const float4*)(H + (size_t)s * HID);
  float4 v = hs[lane];                       // dims 4*lane .. 4*lane+3
  float* ad = agg + (size_t)d * HID + 4 * lane;
  atomicAdd(ad + 0, v.x);
  atomicAdd(ad + 1, v.y);
  atomicAdd(ad + 2, v.z);
  atomicAdd(ad + 3, v.w);
}

// ---------------------------------------------------------------------------
// Fused GEMM + bias + ELU:  Out[r][n] = elu( sum_k A[r][k]*W[k][n] + b[n] )
// A: f32 (converted to f16 on load), WT: f16 transposed, accumulate f32 in
// WMMA. One wave owns a 16-row strip (16x128 output), 8 waves per block.
//
// A fragment (16x32 f16), lane L: M=L%16, half=L/16:
//   elements 0..15 -> K = kbase + half*16 + h (contiguous, same order as B)
// B fragment (32x16 f16), lane L: N=L%16, K = kbase + half*16 + h (contiguous)
// C/D (16x16 f32), VGPR r: row = r + half*8, col = L%16.
// ---------------------------------------------------------------------------
__global__ void gemm_bias_elu_kernel(const float* __restrict__ A,
                                     const _Float16* __restrict__ WT,
                                     const float* __restrict__ bias,
                                     float* __restrict__ Out) {
  const int wave = threadIdx.x >> 5;            // 0..7
  const int lane = threadIdx.x & 31;
  const int m    = lane & 15;
  const int half = lane >> 4;
  const int rowTile = blockIdx.x * 8 + wave;    // 16-row tile index
  const int row0 = rowTile * 16;

  // Load the full K=128 strip of A for this lane's row as 4 fragments.
  const float* arow = A + (size_t)(row0 + m) * HID;
  v16h afrag[4];
#pragma unroll
  for (int kf = 0; kf < 4; ++kf) {
    const int kb = kf * 32 + half * 16;
#pragma unroll
    for (int i = 0; i < 8; ++i) {
      afrag[kf][i]     = (_Float16)arow[kb + i];
      afrag[kf][i + 8] = (_Float16)arow[kb + 8 + i];
    }
  }

#pragma unroll
  for (int nt = 0; nt < 8; ++nt) {
    const int nbase = nt * 16;
    v8f c = {};
#pragma unroll
    for (int kf = 0; kf < 4; ++kf) {
      // contiguous 16 halves: WT[(nbase+m)][kf*32 + half*16 ... +15]
      v16h b = *(const v16h*)(WT + (size_t)(nbase + m) * HID + kf * 32 + half * 16);
      c = __builtin_amdgcn_wmma_f32_16x16x32_f16(
          /*neg_a=*/false, afrag[kf], /*neg_b=*/false, b,
          /*c_mod=*/(short)0, c, /*reuse_a=*/false, /*reuse_b=*/false);
    }
    const int ocol = nbase + m;
    const float bn = bias[ocol];
#pragma unroll
    for (int r = 0; r < 8; ++r) {
      const int orow = row0 + half * 8 + r;
      float v = c[r] + bn;
      Out[(size_t)orow * HID + ocol] = (v > 0.0f) ? v : (__expf(v) - 1.0f);
    }
  }
}

// ---------------------------------------------------------------------------
// Per-graph accumulation: one block per node, 128 threads = dims.
// ---------------------------------------------------------------------------
__global__ void pool_accum_kernel(const float* __restrict__ H,
                                  const int* __restrict__ gid,
                                  float* __restrict__ sums,
                                  float* __restrict__ counts) {
  int node = blockIdx.x;
  int d = threadIdx.x;
  int g = gid[node];
  atomicAdd(sums + g * HID + d, H[(size_t)node * HID + d]);
  if (d == 0) atomicAdd(counts + g, 1.0f);
}

// ---------------------------------------------------------------------------
// Final classifier: out[g][o] = (sums[g]/max(count,1)) . Wc[:,o] + bc[o]
// ---------------------------------------------------------------------------
__global__ void classify_kernel(const float* __restrict__ sums,
                                const float* __restrict__ counts,
                                const float* __restrict__ Wc,
                                const float* __restrict__ bc,
                                float* __restrict__ out) {
  int g = blockIdx.x;
  int o = threadIdx.x;
  if (o >= OUTD) return;
  float inv = 1.0f / fmaxf(counts[g], 1.0f);
  float acc = bc[o];
  const float* sg = sums + g * HID;
#pragma unroll 4
  for (int d = 0; d < HID; ++d) acc += (sg[d] * inv) * Wc[d * OUTD + o];
  out[g * OUTD + o] = acc;
}

// ---------------------------------------------------------------------------
extern "C" void kernel_launch(void* const* d_in, const int* in_sizes, int n_in,
                              void* d_out, int out_size, void* d_ws, size_t ws_size,
                              hipStream_t stream) {
  const float* features = (const float*)d_in[0];
  const int*   src      = (const int*)d_in[1];
  const int*   dst      = (const int*)d_in[2];
  const int*   gids     = (const int*)d_in[3];
  const float* W[3]     = {(const float*)d_in[4], (const float*)d_in[6], (const float*)d_in[8]};
  const float* b[3]     = {(const float*)d_in[5], (const float*)d_in[7], (const float*)d_in[9]};
  const float* Wc       = (const float*)d_in[10];
  const float* bc       = (const float*)d_in[11];
  float* out            = (float*)d_out;

  char* ws = (char*)d_ws;
  const size_t MAT = (size_t)NNODES * HID * sizeof(float);   // 32 MB
  float*    agg   = (float*)(ws);
  float*    h     = (float*)(ws + MAT);
  _Float16* WT[3];
  size_t off = 2 * MAT;
  for (int l = 0; l < 3; ++l) { WT[l] = (_Float16*)(ws + off); off += (size_t)HID * HID * sizeof(_Float16); }
  float* sums   = (float*)(ws + off); off += (size_t)NGRAPHS * HID * sizeof(float);
  float* counts = (float*)(ws + off); off += (size_t)NGRAPHS * sizeof(float);

  // Convert/transpose the three layer weights to f16.
  for (int l = 0; l < 3; ++l)
    convert_wt_kernel<<<HID, HID, 0, stream>>>(W[l], WT[l]);

  const int scatterBlocks = NEDGES / 8;     // 8 waves (edges) per 256-thread block
  const int gemmBlocks    = NNODES / (16 * 8);

  for (int l = 0; l < 3; ++l) {
    hipMemsetAsync(agg, 0, MAT, stream);
    const float* hin = (l == 0) ? features : h;
    scatter_add_kernel<<<scatterBlocks, 256, 0, stream>>>(hin, src, dst, agg);
    gemm_bias_elu_kernel<<<gemmBlocks, 256, 0, stream>>>(agg, WT[l], b[l], h);
  }

  hipMemsetAsync(sums, 0, (size_t)NGRAPHS * HID * sizeof(float) + NGRAPHS * sizeof(float), stream);
  pool_accum_kernel<<<NNODES, HID, 0, stream>>>(h, gids, sums, counts);
  classify_kernel<<<NGRAPHS, 16, 0, stream>>>(sums, counts, Wc, bc, out);
}
